// GRUStyleRefinementAttention_67765993996936
// MI455X (gfx1250) — matrix operand order repaired and checked by reading, not compiled, not run on hardware
//
#include <hip/hip_runtime.h>

typedef __attribute__((ext_vector_type(16))) _Float16 v16h;
typedef __attribute__((ext_vector_type(8)))  _Float16 v8h;
typedef __attribute__((ext_vector_type(8)))  float    v8f;

#define TT 2048
#define CC 768
#define HH 12
#define DD 64
#define RR 32
#define C4 (4*CC)
#define BSCAL (-7.6246189861593985f)   /* -log(2048) */
#define NEG_EPS 1e-30f

// ---------------------------------------------------------------------------
// WMMA helpers (gfx1250 wave32, v_wmma_f32_16x16x32_f16)
// ---------------------------------------------------------------------------
__device__ __forceinline__ v8f wmma_f16(v16h a, v16h b, v8f c) {
  return __builtin_amdgcn_wmma_f32_16x16x32_f16(false, a, false, b, (short)0, c,
                                                false, false);
}

// A fragment: 16x32 f16 tile, row-major source with leading dim ld.
// lane l: m = l&15, hi = l>>4 -> K = 16hi..16hi+15 (contiguous along K)
__device__ __forceinline__ v16h frag_a(const _Float16* base, int ld) {
  const int l  = threadIdx.x & 31;
  const int m  = l & 15;
  const int hi = l >> 4;
  const _Float16* p = base + (size_t)m * ld + hi * 16;
  v8h lo = *(const v8h*)(p);
  v8h hh = *(const v8h*)(p + 8);
  v16h a;
#pragma unroll
  for (int i = 0; i < 8; ++i) { a[i] = lo[i]; a[i + 8] = hh[i]; }
  return a;
}

// B fragment: 32x16 (KxN) from N-major (transposed) storage with leading dim ld.
// lane l: n = l&15, K range = (l>>4)*16 .. +15 (contiguous along K)
__device__ __forceinline__ v16h frag_b(const _Float16* base, int ld) {
  const int l  = threadIdx.x & 31;
  const int n  = l & 15;
  const int kb = (l >> 4) * 16;
  const _Float16* p = base + (size_t)n * ld + kb;
  v8h lo = *(const v8h*)(p);
  v8h hh = *(const v8h*)(p + 8);
  v16h b;
#pragma unroll
  for (int i = 0; i < 8; ++i) { b[i] = lo[i]; b[i + 8] = hh[i]; }
  return b;
}

__device__ __forceinline__ v8f zero8() {
  v8f r;
#pragma unroll
  for (int i = 0; i < 8; ++i) r[i] = 0.f;
  return r;
}

__device__ __forceinline__ float sigf(float x) { return 1.f / (1.f + expf(-x)); }

// ---------------------------------------------------------------------------
// Generic WMMA GEMM: C[M,N] = A[M,K] @ B[K,N] (+bias) ; Bt is B transposed NxK.
// act=1 applies sigmoid. Cf (f32) and Ch (f16) outputs are each optional.
// One wave per 16x16 output tile, K-loop step 32, one wmma per step.
// ---------------------------------------------------------------------------
__global__ void k_gemm16(const _Float16* __restrict__ A,
                         const _Float16* __restrict__ Bt,
                         const float* __restrict__ bias,
                         float* __restrict__ Cf, _Float16* __restrict__ Ch,
                         int M, int N, int K, int act) {
  const int n0 = blockIdx.x * 16;
  const int m0 = blockIdx.y * 16;
  v8f acc = zero8();
  for (int k0 = 0; k0 < K; k0 += 32) {
    v16h a = frag_a(A + (size_t)m0 * K + k0, K);
    v16h b = frag_b(Bt + (size_t)n0 * K + k0, K);
    acc = wmma_f16(a, b, acc);
  }
  const int l = threadIdx.x & 31;
  const int n = l & 15, hi = l >> 4;
  const float bv = bias ? bias[n0 + n] : 0.f;
#pragma unroll
  for (int v = 0; v < 8; ++v) {
    float y = acc[v] + bv;
    if (act) y = sigf(y);
    const size_t idx = (size_t)(m0 + v + 8 * hi) * N + n0 + n;
    if (Cf) Cf[idx] = y;
    if (Ch) Ch[idx] = (_Float16)y;
  }
}

// ---------------------------------------------------------------------------
// Seed attention (low-rank, full causal, R=32):
//   S = (Qr/sqrt(R)) Kr^T ; A = sigmoid(S^2 + b) masked causal
//   Yr = A @ exp(Vr) ; H0pre = log(max(Yr*exp(-m_seed), eps)) + m_seed
// One wave per 16-query tile; S via one wmma (K=R=32); probs restaged through
// LDS into A-layout; PV via two wmmas against exp(Vr) transposed (RxT).
// ---------------------------------------------------------------------------
__global__ void k_seed_attn(const _Float16* __restrict__ Qr16,
                            const _Float16* __restrict__ Kr16,
                            const _Float16* __restrict__ eVT,
                            const float* __restrict__ m_seed,
                            _Float16* __restrict__ H0pre) {
  __shared__ alignas(16) _Float16 Alds[16 * 32];
  const int i0  = blockIdx.x * 16;
  const int l   = threadIdx.x & 31;
  const int lan = l & 15, hi = l >> 4;
  v16h aQ = frag_a(Qr16 + (size_t)i0 * RR, RR);
  v8f y0 = zero8(), y1 = zero8();

  for (int j0 = 0; j0 <= i0; j0 += 32) {
#pragma unroll
    for (int s = 0; s < 2; ++s) {
      const int jt = j0 + s * 16;
      v16h bK = frag_b(Kr16 + (size_t)jt * RR, RR);
      v8f sc = zero8();
      sc = wmma_f16(aQ, bK, sc);
#pragma unroll
      for (int v = 0; v < 8; ++v) {
        const int qrow = i0 + v + 8 * hi;
        const int key  = jt + lan;
        float p = 0.f;
        if (key <= qrow) { const float sv = sc[v]; p = sigf(sv * sv + BSCAL); }
        Alds[(v + 8 * hi) * 32 + s * 16 + lan] = (_Float16)p;
      }
    }
    __syncthreads();
    v16h aP = frag_a(Alds, 32);
    v16h b0 = frag_b(eVT + j0, TT);                       // r-cols 0..15
    v16h b1 = frag_b(eVT + (size_t)16 * TT + j0, TT);     // r-cols 16..31
    y0 = wmma_f16(aP, b0, y0);
    y1 = wmma_f16(aP, b1, y1);
    __syncthreads();
  }
#pragma unroll
  for (int v = 0; v < 8; ++v) {
    const int row = i0 + v + 8 * hi;
    {
      const float m = m_seed[row * RR + lan];
      const float y = y0[v] * expf(-m);
      H0pre[row * RR + lan] = (_Float16)(logf(fmaxf(y, NEG_EPS)) + m);
    }
    {
      const float m = m_seed[row * RR + 16 + lan];
      const float y = y1[v] * expf(-m);
      H0pre[row * RR + 16 + lan] = (_Float16)(logf(fmaxf(y, NEG_EPS)) + m);
    }
  }
}

// ---------------------------------------------------------------------------
// Near-field banded attention + head gating. One wave per (head, 16-q tile).
// cos tile via 2 wmmas (K=D=64); PV 16x64 via 4 wmmas per 32-key chunk.
// ---------------------------------------------------------------------------
__global__ void k_near_attn(const _Float16* __restrict__ Qn,
                            const _Float16* __restrict__ Kn,
                            const _Float16* __restrict__ VeT,
                            const float* __restrict__ m_near,
                            const float* __restrict__ gate_u,
                            const float* __restrict__ gate_b,
                            float* __restrict__ Htil) {
  __shared__ alignas(16) _Float16 Alds[16 * 32];
  __shared__ float hn[16 * 64];
  const int i0  = blockIdx.x * 16;
  const int h   = blockIdx.y;
  const int l   = threadIdx.x & 31;
  const int lan = l & 15, hi = l >> 4;
  const float invtau = 1.0f / 0.35f;

  const _Float16* qb = Qn + (size_t)i0 * CC + h * DD;
  v16h aq0 = frag_a(qb, CC);        // d 0..31
  v16h aq1 = frag_a(qb + 32, CC);   // d 32..63
  v8f acc[4];
#pragma unroll
  for (int i = 0; i < 4; ++i) acc[i] = zero8();

  for (int j0 = i0 - 128; j0 <= i0; j0 += 32) {
    if (j0 + 31 < 0) continue;   // whole chunk below 0
#pragma unroll
    for (int s = 0; s < 2; ++s) {
      const int jt  = j0 + s * 16;
      const bool any = (jt <= i0) && (jt + 15 >= 0);
      v8f sc = zero8();
      if (any) {
        int key = jt + lan; if (key < 0) key = 0;   // clamped, masked below
        const _Float16* kp = Kn + (size_t)key * CC + h * DD + hi * 16;
        v8h l0 = *(const v8h*)(kp);
        v8h h0 = *(const v8h*)(kp + 8);
        v8h l1 = *(const v8h*)(kp + 32);
        v8h h1 = *(const v8h*)(kp + 40);
        v16h bk0, bk1;
#pragma unroll
        for (int i = 0; i < 8; ++i) {
          bk0[i] = l0[i]; bk0[i + 8] = h0[i];
          bk1[i] = l1[i]; bk1[i + 8] = h1[i];
        }
        sc = wmma_f16(aq0, bk0, sc);
        sc = wmma_f16(aq1, bk1, sc);
      }
#pragma unroll
      for (int v = 0; v < 8; ++v) {
        const int qrow = i0 + v + 8 * hi;
        const int key  = jt + lan;
        float p = 0.f;
        if (any && key >= 0 && key <= qrow && (qrow - key) <= 128) {
          const float cv = sc[v];
          p = sigf(cv * cv * invtau + BSCAL);
        }
        Alds[(v + 8 * hi) * 32 + s * 16 + lan] = (_Float16)p;
      }
    }
    __syncthreads();
    v16h aP = frag_a(Alds, 32);
    int col0 = j0 + hi * 16;           // per-half key base, clamp (masked by A=0)
    if (col0 < 0) col0 = 0;
    if (col0 > TT - 16) col0 = TT - 16;
#pragma unroll
    for (int nh = 0; nh < 4; ++nh) {
      const int d = nh * 16 + lan;
      const _Float16* vp = VeT + (size_t)(h * DD + d) * TT + col0;
      v8h lo = *(const v8h*)(vp);
      v8h hh = *(const v8h*)(vp + 8);
      v16h bv;
#pragma unroll
      for (int i = 0; i < 8; ++i) { bv[i] = lo[i]; bv[i + 8] = hh[i]; }
      acc[nh] = wmma_f16(aP, bv, acc[nh]);
    }
    __syncthreads();
  }

  // H_near = log(max(Y,eps)) + m_near, staged in LDS for head gating
#pragma unroll
  for (int nh = 0; nh < 4; ++nh) {
#pragma unroll
    for (int v = 0; v < 8; ++v) {
      const int m  = v + 8 * hi;
      const int d  = nh * 16 + lan;
      const float mn = m_near[(size_t)(i0 + m) * CC + h * DD + d];
      hn[m * 64 + d] = logf(fmaxf(acc[nh][v], NEG_EPS)) + mn;
    }
  }
  __syncthreads();
  if (l < 16) {
    const int row = l;
    float s = 0.f;
    for (int d = 0; d < DD; ++d) s += hn[row * 64 + d] * gate_u[h * DD + d];
    const float g = sigf(s + gate_b[h]);
    for (int d = 0; d < DD; ++d)
      Htil[(size_t)(i0 + row) * CC + h * DD + d] = hn[row * 64 + d] * g;
  }
}

// ---------------------------------------------------------------------------
// Elementwise / prep kernels
// ---------------------------------------------------------------------------
__global__ void k_cvt(const float* __restrict__ in, _Float16* __restrict__ out,
                      int n, float scale) {
  const int i = blockIdx.x * blockDim.x + threadIdx.x;
  if (i < n) out[i] = (_Float16)(in[i] * scale);
}

// out[fout, fin] = in[fin, fout]^T  (f32 -> f16)
__global__ void k_cvtT(const float* __restrict__ in, _Float16* __restrict__ out,
                       int fin, int fout) {
  const int i = blockIdx.x * blockDim.x + threadIdx.x;
  if (i >= fin * fout) return;
  const int r = i / fout, c = i - r * fout;
  out[(size_t)c * fin + r] = (_Float16)in[i];
}

// m_seed = cummax(Vr) along T ; eVT[r, t] = exp(Vr[t, r])
__global__ void k_cummax(const float* __restrict__ Vr,
                         float* __restrict__ m_seed,
                         _Float16* __restrict__ eVT) {
  const int r = threadIdx.x;   // 32 lanes, one per rank column
  float run = -1e30f;
  for (int t = 0; t < TT; ++t) {
    const float v = Vr[t * RR + r];
    run = fmaxf(run, v);
    m_seed[t * RR + r] = run;
    eVT[(size_t)r * TT + t] = (_Float16)expf(v);
  }
}

// per-head L2 row normalization, f32 -> f16
__global__ void k_l2n(const float* __restrict__ X, _Float16* __restrict__ Xn) {
  const int idx = blockIdx.x * blockDim.x + threadIdx.x;   // T*H
  if (idx >= TT * HH) return;
  const int t = idx / HH;
  const int h = idx - t * HH;
  const float* p = X + (size_t)t * CC + h * DD;
  float s = 0.f;
  for (int d = 0; d < DD; ++d) s += p[d] * p[d];
  const float inv = 1.f / fmaxf(sqrtf(s), 1e-12f);
  _Float16* o = Xn + (size_t)t * CC + h * DD;
  for (int d = 0; d < DD; ++d) o[d] = (_Float16)(p[d] * inv);
}

// sliding past-window (width 129) max of V ; VeT[c, t] = exp(V[t,c] - m)
__global__ void k_mnear(const float* __restrict__ V, float* __restrict__ m_near,
                        _Float16* __restrict__ VeT) {
  const int idx = blockIdx.x * blockDim.x + threadIdx.x;
  if (idx >= TT * CC) return;
  const int t = idx / CC;
  const int c = idx - t * CC;
  int lo = t - 128; if (lo < 0) lo = 0;
  float m = -1e30f;
  for (int s = lo; s <= t; ++s) m = fmaxf(m, V[(size_t)s * CC + c]);
  m_near[idx] = m;
  VeT[(size_t)c * TT + t] = (_Float16)expf(V[idx] - m);
}

// zin = concat([x, H0, Htilde, H0*Htilde]) in f16, row stride 4C
__global__ void k_zin(const float* __restrict__ x, const float* __restrict__ H0,
                      const float* __restrict__ Ht, _Float16* __restrict__ zin) {
  const int i = blockIdx.x * blockDim.x + threadIdx.x;
  if (i >= TT * CC) return;
  const int t = i / CC, c = i - t * CC;
  const float a = x[i], b = H0[i], d = Ht[i];
  _Float16* r = zin + (size_t)t * C4;
  r[c]          = (_Float16)a;
  r[CC + c]     = (_Float16)b;
  r[2 * CC + c] = (_Float16)d;
  r[3 * CC + c] = (_Float16)(b * d);
}

// Hout16 = (1-z)*H0 + z*Htilde  (f16 for final GEMM)
__global__ void k_blend(const float* __restrict__ z, const float* __restrict__ H0,
                        const float* __restrict__ Ht, _Float16* __restrict__ out16) {
  const int i = blockIdx.x * blockDim.x + threadIdx.x;
  if (i >= TT * CC) return;
  const float zz = z[i];
  out16[i] = (_Float16)((1.f - zz) * H0[i] + zz * Ht[i]);
}

// ---------------------------------------------------------------------------
// Host orchestration
// ---------------------------------------------------------------------------
extern "C" void kernel_launch(void* const* d_in, const int* in_sizes, int n_in,
                              void* d_out, int out_size, void* d_ws, size_t ws_size,
                              hipStream_t stream) {
  (void)in_sizes; (void)n_in; (void)out_size; (void)ws_size;
  const float* x      = (const float*)d_in[0];
  const float* Wq_w   = (const float*)d_in[1];
  const float* Wq_b   = (const float*)d_in[2];
  const float* Wk_w   = (const float*)d_in[3];
  const float* Wk_b   = (const float*)d_in[4];
  const float* Wv_w   = (const float*)d_in[5];
  const float* Wv_b   = (const float*)d_in[6];
  const float* Wo_w   = (const float*)d_in[7];
  const float* Wo_b   = (const float*)d_in[8];
  const float* Pq_w   = (const float*)d_in[9];
  const float* Pk_w   = (const float*)d_in[10];
  const float* Pv_w   = (const float*)d_in[11];
  const float* up_w   = (const float*)d_in[12];
  const float* gate_u = (const float*)d_in[13];
  const float* gate_b = (const float*)d_in[14];
  const float* Wz_w   = (const float*)d_in[15];
  const float* Wz_b   = (const float*)d_in[16];
  float* out = (float*)d_out;

  char* w = (char*)d_ws;
  size_t off = 0;
  auto alloc = [&](size_t bytes) {
    void* p = w + off;
    off = (off + bytes + 255) & ~(size_t)255;
    return p;
  };
  const size_t TC = (size_t)TT * CC;

  // f16 scratch
  _Float16* xh     = (_Float16*)alloc(TC * 2);
  _Float16* Wqt    = (_Float16*)alloc((size_t)CC * CC * 2);
  _Float16* Wkt    = (_Float16*)alloc((size_t)CC * CC * 2);
  _Float16* Wvt    = (_Float16*)alloc((size_t)CC * CC * 2);
  _Float16* Wot    = (_Float16*)alloc((size_t)CC * CC * 2);
  _Float16* Wzt    = (_Float16*)alloc((size_t)CC * C4 * 2);
  _Float16* Pqt    = (_Float16*)alloc((size_t)RR * CC * 2);
  _Float16* Pkt    = (_Float16*)alloc((size_t)RR * CC * 2);
  _Float16* Pvt    = (_Float16*)alloc((size_t)RR * CC * 2);
  _Float16* upT    = (_Float16*)alloc((size_t)CC * RR * 2);
  _Float16* Q16    = (_Float16*)alloc(TC * 2);
  _Float16* K16    = (_Float16*)alloc(TC * 2);
  _Float16* V16    = (_Float16*)alloc(TC * 2);
  _Float16* Qr16   = (_Float16*)alloc((size_t)TT * RR * 2);
  _Float16* Kr16   = (_Float16*)alloc((size_t)TT * RR * 2);
  _Float16* eVT    = (_Float16*)alloc((size_t)RR * TT * 2);
  _Float16* H0pre  = (_Float16*)alloc((size_t)TT * RR * 2);
  _Float16* Qn16   = (_Float16*)alloc(TC * 2);
  _Float16* Kn16   = (_Float16*)alloc(TC * 2);
  _Float16* VeT16  = (_Float16*)alloc(TC * 2);
  _Float16* zin16  = (_Float16*)alloc((size_t)TT * C4 * 2);
  _Float16* Hout16 = (_Float16*)alloc(TC * 2);
  // f32 scratch
  float* Qf     = (float*)alloc(TC * 4);
  float* Kf     = (float*)alloc(TC * 4);
  float* Vf     = (float*)alloc(TC * 4);
  float* Qrf    = (float*)alloc((size_t)TT * RR * 4);
  float* Vrf    = (float*)alloc((size_t)TT * RR * 4);
  float* m_seed = (float*)alloc((size_t)TT * RR * 4);
  float* H0f    = (float*)alloc(TC * 4);
  float* m_nr   = (float*)alloc(TC * 4);
  float* Htil   = (float*)alloc(TC * 4);
  float* zbuf   = (float*)alloc(TC * 4);

  const int TPB = 256;
  auto nb = [](size_t n, int tpb) { return (unsigned)((n + tpb - 1) / tpb); };

  // ---- convert activations / weights to WMMA-friendly f16 ----
  k_cvt <<<nb(TC, TPB), TPB, 0, stream>>>(x, xh, (int)TC, 1.f);
  k_cvtT<<<nb((size_t)CC * CC, TPB), TPB, 0, stream>>>(Wq_w, Wqt, CC, CC);
  k_cvtT<<<nb((size_t)CC * CC, TPB), TPB, 0, stream>>>(Wk_w, Wkt, CC, CC);
  k_cvtT<<<nb((size_t)CC * CC, TPB), TPB, 0, stream>>>(Wv_w, Wvt, CC, CC);
  k_cvtT<<<nb((size_t)CC * CC, TPB), TPB, 0, stream>>>(Wo_w, Wot, CC, CC);
  k_cvtT<<<nb((size_t)C4 * CC, TPB), TPB, 0, stream>>>(Wz_w, Wzt, C4, CC);
  k_cvtT<<<nb((size_t)CC * RR, TPB), TPB, 0, stream>>>(Pq_w, Pqt, CC, RR);
  k_cvtT<<<nb((size_t)CC * RR, TPB), TPB, 0, stream>>>(Pk_w, Pkt, CC, RR);
  k_cvtT<<<nb((size_t)CC * RR, TPB), TPB, 0, stream>>>(Pv_w, Pvt, CC, RR);
  k_cvtT<<<nb((size_t)RR * CC, TPB), TPB, 0, stream>>>(up_w, upT, RR, CC);

  // ---- QKV projections (WMMA) ----
  dim3 gTC(CC / 16, TT / 16);
  k_gemm16<<<gTC, 32, 0, stream>>>(xh, Wqt, Wq_b, Qf, Q16, TT, CC, CC, 0);
  k_gemm16<<<gTC, 32, 0, stream>>>(xh, Wkt, Wk_b, Kf, K16, TT, CC, CC, 0);
  k_gemm16<<<gTC, 32, 0, stream>>>(xh, Wvt, Wv_b, Vf, V16, TT, CC, CC, 0);

  // ---- low-rank projections (WMMA, N=32) ----
  dim3 gTR(RR / 16, TT / 16);
  k_gemm16<<<gTR, 32, 0, stream>>>(Q16, Pqt, nullptr, Qrf, nullptr, TT, RR, CC, 0);
  k_gemm16<<<gTR, 32, 0, stream>>>(K16, Pkt, nullptr, nullptr, Kr16, TT, RR, CC, 0);
  k_gemm16<<<gTR, 32, 0, stream>>>(V16, Pvt, nullptr, Vrf, nullptr, TT, RR, CC, 0);
  // fold 1/sqrt(R) into the Q-side operand
  k_cvt<<<nb((size_t)TT * RR, TPB), TPB, 0, stream>>>(Qrf, Qr16, TT * RR,
                                                     0.17677669529663687f);
  k_cummax<<<1, 32, 0, stream>>>(Vrf, m_seed, eVT);

  // ---- seed attention + up-projection ----
  k_seed_attn<<<TT / 16, 32, 0, stream>>>(Qr16, Kr16, eVT, m_seed, H0pre);
  k_gemm16<<<gTC, 32, 0, stream>>>(H0pre, upT, nullptr, H0f, nullptr, TT, CC, RR, 0);

  // ---- near-field banded attention ----
  k_l2n  <<<nb((size_t)TT * HH, TPB), TPB, 0, stream>>>(Qf, Qn16);
  k_l2n  <<<nb((size_t)TT * HH, TPB), TPB, 0, stream>>>(Kf, Kn16);
  k_mnear<<<nb(TC, TPB), TPB, 0, stream>>>(Vf, m_nr, VeT16);
  dim3 gNear(TT / 16, HH);
  k_near_attn<<<gNear, 32, 0, stream>>>(Qn16, Kn16, VeT16, m_nr, gate_u, gate_b,
                                        Htil);

  // ---- GRU blend + output projection ----
  k_zin<<<nb(TC, TPB), TPB, 0, stream>>>(x, H0f, Htil, zin16);
  k_gemm16<<<gTC, 32, 0, stream>>>(zin16, Wzt, Wz_b, zbuf, nullptr, TT, CC, C4, 1);
  k_blend<<<nb(TC, TPB), TPB, 0, stream>>>(zbuf, H0f, Htil, Hout16);
  k_gemm16<<<gTC, 32, 0, stream>>>(Hout16, Wot, Wo_b, out, nullptr, TT, CC, CC, 0);
}
